// H2GCN_60653528154338
// MI455X (gfx1250) — matrix operand order repaired and checked by reading, not compiled, not run on hardware
//
#include <hip/hip_runtime.h>
#include <hip/hip_bf16.h>
#include <stdint.h>

// ---------------------------------------------------------------------------
// H2GCN forward on MI455X (gfx1250, wave32, WMMA + TDM).
//   - Gram/distance matrix: v_wmma_f32_16x16x4_f32 (fp32 precision for top-k)
//   - Dense GEMMs (ego/msg/fc1/fc2): LDS-tiled v_wmma_f32_16x16x32_bf16,
//     tiles staged by the Tensor Data Mover (tensor_load_to_lds + tensorcnt)
//   - kNN->symmetrize->2-hop adjacency kept as 256-bit row bitmasks
//   - A2 @ msg done as bitmask-driven gather-sum (sparse, ~8-40 nbrs/row)
// ---------------------------------------------------------------------------

#define BB   128            // batch
#define NN   197            // nodes
#define FF   768            // input features
#define NH   512            // hidden
#define NC   1000           // classes
#define NP   208            // padded N (13 tiles of 16)
#define MM   (BB * NN)      // 25216 rows (= 197 * 128, multiple of 128)
#define MW   8              // mask words per row (256 bits >= 197)

#define BM   128            // GEMM block tile rows
#define BN   64             // GEMM block tile cols
#define BK   32             // GEMM K step (bf16 WMMA depth)

typedef __attribute__((ext_vector_type(2)))  float        v2f;
typedef __attribute__((ext_vector_type(8)))  float        v8f;
typedef __attribute__((ext_vector_type(16))) __bf16       v16bf;
typedef __attribute__((ext_vector_type(4)))  unsigned int v4u;
typedef __attribute__((ext_vector_type(8)))  int          v8i;
typedef __attribute__((ext_vector_type(4)))  int          v4i;

// ----------------------------- utility kernels -----------------------------

__global__ void k_zero_u32(uint32_t* __restrict__ p, int n) {
  int i = blockIdx.x * blockDim.x + threadIdx.x;
  if (i < n) p[i] = 0u;
}

__global__ void k_f32_to_bf16(const float* __restrict__ src, __bf16* __restrict__ dst, int n) {
  int i = blockIdx.x * blockDim.x + threadIdx.x;
  if (i < n) dst[i] = (__bf16)src[i];
}

__global__ void k_sqnorm(const float* __restrict__ X, float* __restrict__ sq) {
  int t = blockIdx.x * blockDim.x + threadIdx.x;
  if (t >= MM) return;
  const float* row = X + (size_t)t * FF;
  float acc = 0.f;
  for (int k = 0; k < FF; k += 4) {
    float4 v = *(const float4*)(row + k);
    acc += v.x * v.x + v.y * v.y + v.z * v.z + v.w * v.w;
  }
  sq[t] = acc;
}

// --------------------- Gram / distance matrix (f32 WMMA) --------------------
// dist[b, r, c] = |x_r|^2 + |x_c|^2 - 2 <x_r, x_c>, diag & padding = +INF.

__global__ void k_gram_dist(const float* __restrict__ X, const float* __restrict__ sq,
                            float* __restrict__ dist) {
  const int lane = threadIdx.x & 31;
  const int tn = blockIdx.x, tm = blockIdx.y, b = blockIdx.z;

  int ar = tm * 16 + (lane & 15); if (ar >= NN) ar = NN - 1;   // clamped load row
  int br = tn * 16 + (lane & 15); if (br >= NN) br = NN - 1;
  const float* arow = X + ((size_t)b * NN + ar) * FF;
  const float* brow = X + ((size_t)b * NN + br) * FF;
  const int koff = (lane >> 4) * 2;        // lanes 16-31 hold K+2,K+3

  v8f acc = {};
  for (int k = 0; k < FF; k += 4) {
    v2f a  = *(const v2f*)(arow + k + koff);
    v2f bf = *(const v2f*)(brow + k + koff);
    acc = __builtin_amdgcn_wmma_f32_16x16x4_f32(false, a, false, bf,
                                                (short)0, acc, false, false);
  }

  const int col = tn * 16 + (lane & 15);
  const float sc = sq[(size_t)b * NN + ((col < NN) ? col : (NN - 1))];
#pragma unroll
  for (int v = 0; v < 8; ++v) {
    const int row = tm * 16 + v + ((lane >> 4) << 3);
    float d;
    if (col >= NN || row >= NN || row == col) d = 3.0e38f;
    else d = sq[(size_t)b * NN + row] + sc - 2.0f * acc[v];
    dist[((size_t)b * NP + row) * NP + col] = d;
  }
}

// --------------------------- top-k=4 -> 1-hop mask --------------------------

__global__ void k_topk(float* __restrict__ dist, uint32_t* __restrict__ Amask) {
  int t = blockIdx.x * blockDim.x + threadIdx.x;
  if (t >= MM) return;
  const int b = t / NN, n = t % NN;
  float* row = dist + ((size_t)b * NP + n) * NP;
#pragma unroll 1
  for (int j = 0; j < 4; ++j) {
    float best = 3.0e38f; int bi = 0;
    for (int m = 0; m < NN; ++m) {
      float d = row[m];
      if (d < best) { best = d; bi = m; }
    }
    row[bi] = 3.0e38f;
    atomicOr(&Amask[(size_t)t * MW + (bi >> 5)], 1u << (bi & 31));                 // A[n,bi]
    atomicOr(&Amask[((size_t)b * NN + bi) * MW + (n >> 5)], 1u << (n & 31));       // A[bi,n]
  }
}

// ------------------- 2-hop adjacency (binary) + degree ----------------------

__global__ void k_twohop(const uint32_t* __restrict__ Amask, uint32_t* __restrict__ A2,
                         float* __restrict__ deg) {
  int t = blockIdx.x * blockDim.x + threadIdx.x;
  if (t >= MM) return;
  const int b = t / NN, n = t % NN;
  uint32_t out[MW];
#pragma unroll
  for (int q = 0; q < MW; ++q) out[q] = 0u;
  const uint32_t* own = Amask + (size_t)t * MW;
  for (int w = 0; w < 7; ++w) {            // 197 bits < 7*32
    uint32_t bits = own[w];
    while (bits) {
      const int m = w * 32 + __ffs(bits) - 1;
      bits &= bits - 1;
      const uint32_t* nm = Amask + ((size_t)b * NN + m) * MW;
#pragma unroll
      for (int q = 0; q < MW; ++q) out[q] |= nm[q];
    }
  }
  out[n >> 5] &= ~(1u << (n & 31));        // fill_diagonal_(0)
  int cnt = 0;
#pragma unroll
  for (int q = 0; q < MW; ++q) { A2[(size_t)t * MW + q] = out[q]; cnt += __popc(out[q]); }
  deg[t] = (cnt > 0) ? (float)cnt : 1.0f;
}

// ------------------ TDM: 2-D bf16 tile  global -> LDS -----------------------
// D# per CDNA5 ISA 8.3/8.4. global_addr = tile start; tensor dims are the
// remaining extent from the tile start so OOB rows/cols read back as zero.
// Issued by one wave (TDM ignores EXEC; descriptor operands are uniform).
// This toolchain's builtin takes 6 args: (g0, g1, g2, g3, g_extra, cpol).

__device__ static inline void tdm_load_2d(unsigned lds_addr, const __bf16* gptr,
                                          int width_elems, int rows_left,
                                          int tile_w, int tile_h, int row_stride_elems) {
  unsigned long long ga = (unsigned long long)(size_t)gptr;
  v4u g0;
  g0.x = 1u;                                               // count=1, user mode
  g0.y = lds_addr;                                         // [63:32] lds_addr
  g0.z = (unsigned)(ga & 0xffffffffu);                     // [95:64] global_addr lo
  g0.w = (unsigned)((ga >> 32) & 0x01ffffffu) | (2u << 30);// [120:96] ga hi, type=2

  unsigned d0 = (unsigned)width_elems;                     // tensor_dim0
  unsigned d1 = (unsigned)rows_left;                       // tensor_dim1
  unsigned long long s0 = (unsigned long long)row_stride_elems; // dim0 stride
  v8i g1;
  g1[0] = (int)(1u << 16);                                 // mask=0, data_size=1 (2B)
  g1[1] = (int)((d0 & 0xffffu) << 16);                     // [63:48] dim0 lo16
  g1[2] = (int)((d0 >> 16) | ((d1 & 0xffffu) << 16));      // [79:64] dim0 hi, [95:80] dim1 lo
  g1[3] = (int)((d1 >> 16) | ((unsigned)tile_w << 16));    // [111:96] dim1 hi, [127:112] tile_dim0
  g1[4] = (int)((unsigned)tile_h);                         // [143:128] tile_dim1, tile_dim2=0
  g1[5] = (int)(s0 & 0xffffffffu);                         // [191:160] dim0_stride lo
  g1[6] = (int)((s0 >> 32) & 0xffffu);                     // [207:192] dim0_stride hi
  g1[7] = 0;                                               // dim1_stride (unused, 2-D tile)
  v4i z4 = {0, 0, 0, 0};
  v8i z8 = {0, 0, 0, 0, 0, 0, 0, 0};
  __builtin_amdgcn_tensor_load_to_lds(g0, g1, z4, z4, z8, 0);
}

// ------------------------- bf16 WMMA GEMM:  Y = X W^T + b -------------------
// X: [Mr,K] bf16 row-major, W: [Nv,K] bf16 row-major (W rows == B columns).
// Block = 256 thr (8 waves) -> BM x BN tile; wave (wm,wn) owns 2x2 16x16
// subtiles. K multiple of 32; Mr multiple of BM (or == BM); ragged Nv handled
// by TDM OOB-zero + store guard.

__global__ void k_gemm_bf16(const __bf16* __restrict__ X, const __bf16* __restrict__ W,
                            const float* __restrict__ bias, float* __restrict__ Y,
                            int Mr, int Nv, int K, int ldY, int relu) {
  __shared__ __bf16 ldsX[BM * BK];          // 8 KB
  __shared__ __bf16 ldsW[BN * BK];          // 4 KB

  const int tid  = threadIdx.x;
  const int lane = tid & 31;
  const int wm   = (tid >> 5) & 3;          // 4 row groups of 32
  const int wn   = tid >> 7;                // 2 col groups of 32
  const int row0 = blockIdx.y * BM;
  const int col0 = blockIdx.x * BN;

  const unsigned ldsXa = (unsigned)(size_t)&ldsX[0];   // low 32 bits = LDS offset
  const unsigned ldsWa = (unsigned)(size_t)&ldsW[0];

  v8f acc00 = {}, acc01 = {}, acc10 = {}, acc11 = {};

  const int aoff = (lane >> 4) * 8;         // A frag: K chunks {kb..kb+7, +16}
  const int boff = (lane >> 4) * 8;         // B frag: K chunks {kb..kb+7, +16}

  for (int k = 0; k < K; k += BK) {
    if (tid < 32) {                         // wave 0 drives the TDM
      tdm_load_2d(ldsXa, X + (size_t)row0 * K + k, K - k, Mr - row0, BK, BM, K);
      tdm_load_2d(ldsWa, W + (size_t)col0 * K + k, K - k, Nv - col0, BK, BN, K);
      __builtin_amdgcn_s_wait_tensorcnt(0);
    }
    __syncthreads();

    union { v16bf v; uint4 q[2]; } a0, a1, b0, b1;
    const __bf16* ax0 = &ldsX[(wm * 32 + (lane & 15)) * BK];
    const __bf16* ax1 = ax0 + 16 * BK;
    a0.q[0] = *(const uint4*)(ax0 + aoff);
    a0.q[1] = *(const uint4*)(ax0 + aoff + 16);
    a1.q[0] = *(const uint4*)(ax1 + aoff);
    a1.q[1] = *(const uint4*)(ax1 + aoff + 16);
    const __bf16* bw0 = &ldsW[(wn * 32 + (lane & 15)) * BK];
    const __bf16* bw1 = bw0 + 16 * BK;
    b0.q[0] = *(const uint4*)(bw0 + boff);
    b0.q[1] = *(const uint4*)(bw0 + boff + 16);
    b1.q[0] = *(const uint4*)(bw1 + boff);
    b1.q[1] = *(const uint4*)(bw1 + boff + 16);

    acc00 = __builtin_amdgcn_wmma_f32_16x16x32_bf16(false, a0.v, false, b0.v, (short)0, acc00, false, false);
    acc01 = __builtin_amdgcn_wmma_f32_16x16x32_bf16(false, a0.v, false, b1.v, (short)0, acc01, false, false);
    acc10 = __builtin_amdgcn_wmma_f32_16x16x32_bf16(false, a1.v, false, b0.v, (short)0, acc10, false, false);
    acc11 = __builtin_amdgcn_wmma_f32_16x16x32_bf16(false, a1.v, false, b1.v, (short)0, acc11, false, false);

    __syncthreads();
  }

  // ---- epilogue: bias (+relu), guarded store --------------------------------
  v8f* accp[2][2] = { { &acc00, &acc01 }, { &acc10, &acc11 } };
#pragma unroll
  for (int i = 0; i < 2; ++i) {
#pragma unroll
    for (int j = 0; j < 2; ++j) {
      const int colb = col0 + wn * 32 + j * 16 + (lane & 15);
      if (colb >= Nv) continue;
      const float bv = bias[colb];
      const v8f av = *accp[i][j];
#pragma unroll
      for (int v = 0; v < 8; ++v) {
        const int row = row0 + wm * 32 + i * 16 + v + ((lane >> 4) << 3);
        if (row < Mr) {
          float val = av[v] + bv;
          if (relu) val = val > 0.f ? val : 0.f;
          Y[(size_t)row * ldY + colb] = val;
        }
      }
    }
  }
}

// ---------- sparse aggregation: h[:,512:1024]=relu(A2@msg/deg), relu ego ----

__global__ void k_agg_relu(float* __restrict__ h, const float* __restrict__ msg,
                           const uint32_t* __restrict__ A2, const float* __restrict__ deg) {
  const int node = blockIdx.x;         // 0..MM-1
  const int b = node / NN;
  const int c = threadIdx.x;           // 256 threads, 2 channels each
  float a0 = 0.f, a1 = 0.f;
  const uint32_t* mask = A2 + (size_t)node * MW;
  for (int w = 0; w < 7; ++w) {
    uint32_t bits = mask[w];
    while (bits) {
      const int m = w * 32 + __ffs(bits) - 1;
      bits &= bits - 1;
      const float* mrow = msg + ((size_t)b * NN + m) * NH;
      a0 += mrow[c];
      a1 += mrow[c + 256];
    }
  }
  const float rd = 1.0f / deg[node];
  float* hrow = h + (size_t)node * (2 * NH);
  hrow[NH + c]       = fmaxf(a0 * rd, 0.f);
  hrow[NH + c + 256] = fmaxf(a1 * rd, 0.f);
  hrow[c]       = fmaxf(hrow[c], 0.f);         // relu(ego), stored raw by GEMM
  hrow[c + 256] = fmaxf(hrow[c + 256], 0.f);
}

// --------------------- global mean pool over nodes -> gf --------------------

__global__ void k_meanpool(const float* __restrict__ h1, const float* __restrict__ h2,
                           float* __restrict__ gf) {
  const int c = blockIdx.x * blockDim.x + threadIdx.x;   // 0..2047
  const int b = blockIdx.y;
  const float* src = (c < 2 * NH) ? h1 : h2;
  const int cc = (c < 2 * NH) ? c : (c - 2 * NH);
  float acc = 0.f;
  for (int n = 0; n < NN; ++n)
    acc += src[((size_t)b * NN + n) * (2 * NH) + cc];
  gf[(size_t)b * (4 * NH) + c] = acc * (1.0f / (float)NN);
}

// ------------------------------- launcher -----------------------------------

extern "C" void kernel_launch(void* const* d_in, const int* in_sizes, int n_in,
                              void* d_out, int out_size, void* d_ws, size_t ws_size,
                              hipStream_t stream) {
  const float* Fet   = (const float*)d_in[0];
  const float* w1_0  = (const float*)d_in[1];
  const float* b1_0  = (const float*)d_in[2];
  const float* w2_0  = (const float*)d_in[3];
  const float* b2_0  = (const float*)d_in[4];
  const float* w1_1  = (const float*)d_in[5];
  const float* b1_1  = (const float*)d_in[6];
  const float* w2_1  = (const float*)d_in[7];
  const float* b2_1  = (const float*)d_in[8];
  const float* fc1_w = (const float*)d_in[9];
  const float* fc1_b = (const float*)d_in[10];
  const float* fc2_w = (const float*)d_in[11];
  const float* fc2_b = (const float*)d_in[12];

  float* out = (float*)d_out;                  // [128,1000] then gf [128,2048]
  float* gf  = out + (size_t)BB * NC;

  // workspace carve-out (256B aligned)
  char* cur = (char*)d_ws;
  auto carve = [&](size_t bytes) -> void* {
    void* p = (void*)cur;
    cur += (bytes + 255) & ~(size_t)255;
    return p;
  };
  float*    dist  = (float*)   carve((size_t)BB * NP * NP * 4);        // 22.2 MB
  float*    sq    = (float*)   carve((size_t)MM * 4);
  uint32_t* Amask = (uint32_t*)carve((size_t)MM * MW * 4);
  uint32_t* A2    = (uint32_t*)carve((size_t)MM * MW * 4);
  float*    deg   = (float*)   carve((size_t)MM * 4);
  float*    h1    = (float*)   carve((size_t)MM * 2 * NH * 4);         // 103 MB
  float*    h2    = (float*)   carve((size_t)MM * 2 * NH * 4);         // 103 MB
  float*    msg   = (float*)   carve((size_t)MM * NH * 4);             // 52 MB
  __bf16*   xbf   = (__bf16*)  carve((size_t)MM * 2 * NH * 2);         // 52 MB (Fet bf16 / h bf16)
  __bf16*   wbf   = (__bf16*)  carve((size_t)1024 * 2048 * 2);         // weight staging
  float*    pbuf  = (float*)   carve((size_t)BB * NH * 4);
  __bf16*   gfbf  = (__bf16*)  carve((size_t)BB * 4 * NH * 2);
  __bf16*   pbf   = (__bf16*)  carve((size_t)BB * NH * 2);
  (void)ws_size; (void)n_in; (void)in_sizes; (void)out_size;

  const int TPB = 256;
  auto cvt = [&](const float* s, __bf16* d, int n) {
    k_f32_to_bf16<<<(n + TPB - 1) / TPB, TPB, 0, stream>>>(s, d, n);
  };

  // ---- adjacency path -------------------------------------------------------
  k_zero_u32<<<(MM * MW + TPB - 1) / TPB, TPB, 0, stream>>>(Amask, MM * MW);
  k_sqnorm<<<(MM + TPB - 1) / TPB, TPB, 0, stream>>>(Fet, sq);
  k_gram_dist<<<dim3(NP / 16, NP / 16, BB), 32, 0, stream>>>(Fet, sq, dist);
  k_topk<<<(MM + TPB - 1) / TPB, TPB, 0, stream>>>(dist, Amask);
  k_twohop<<<(MM + TPB - 1) / TPB, TPB, 0, stream>>>(Amask, A2, deg);

  // ---- layer 0:  in=Fet[25216,768] ------------------------------------------
  cvt(Fet, xbf, MM * FF);
  cvt(w1_0, wbf, NH * FF);
  k_gemm_bf16<<<dim3(NH / BN, MM / BM), 256, 0, stream>>>(xbf, wbf, b1_0, h1,
      MM, NH, FF, 2 * NH, 0);                                           // ego -> h1[:, :512]
  cvt(w2_0, wbf, NH * FF);
  k_gemm_bf16<<<dim3(NH / BN, MM / BM), 256, 0, stream>>>(xbf, wbf, b2_0, msg,
      MM, NH, FF, NH, 0);                                               // msg
  k_agg_relu<<<MM, 256, 0, stream>>>(h1, msg, A2, deg);                 // h1 complete

  // ---- layer 1:  in=h1[25216,1024] ------------------------------------------
  cvt(h1, xbf, MM * 2 * NH);
  cvt(w1_1, wbf, NH * 2 * NH);
  k_gemm_bf16<<<dim3(NH / BN, MM / BM), 256, 0, stream>>>(xbf, wbf, b1_1, h2,
      MM, NH, 2 * NH, 2 * NH, 0);
  cvt(w2_1, wbf, NH * 2 * NH);
  k_gemm_bf16<<<dim3(NH / BN, MM / BM), 256, 0, stream>>>(xbf, wbf, b2_1, msg,
      MM, NH, 2 * NH, NH, 0);
  k_agg_relu<<<MM, 256, 0, stream>>>(h2, msg, A2, deg);                 // h2 complete

  // ---- pooling + head -------------------------------------------------------
  k_meanpool<<<dim3((4 * NH) / TPB, BB), TPB, 0, stream>>>(h1, h2, gf); // gf -> d_out tail
  cvt(gf, gfbf, BB * 4 * NH);
  cvt(fc1_w, wbf, NH * 4 * NH);
  k_gemm_bf16<<<dim3(NH / BN, BB / BM), 256, 0, stream>>>(gfbf, wbf, fc1_b, pbuf,
      BB, NH, 4 * NH, NH, 1);                                           // relu
  cvt(pbuf, pbf, BB * NH);
  cvt(fc2_w, wbf, NC * NH);
  k_gemm_bf16<<<dim3((NC + BN - 1) / BN, BB / BM), 256, 0, stream>>>(pbf, wbf, fc2_b, out,
      BB, NC, NH, NC, 0);                                               // logits -> d_out head
}
